// MTCNN_4252017623304
// MI455X (gfx1250) — matrix operand order repaired and checked by reading, not compiled, not run on hardware
//
#include <hip/hip_runtime.h>

typedef _Float16 h16;
typedef __attribute__((ext_vector_type(16))) _Float16 v16h;
typedef __attribute__((ext_vector_type(8)))  _Float16 v8h;
typedef __attribute__((ext_vector_type(4)))  _Float16 v4h;
typedef __attribute__((ext_vector_type(8)))  float    v8f;

// ---------------------------------------------------------------------------
// Generic implicit-GEMM conv / dense layer using V_WMMA_F32_16X16X32_F16.
// Activations are f16 NHWC; GEMM K dimension ordered (ky,kx,c), c innermost.
// M = N*OH*OW output pixels, cols = output channels, K = KH*KW*C.
// Weights pre-packed f16 [Opad][Kpad] (B^T, zero-padded) in the same K order.
// A-fragment paths:
//   contig: 1x1 conv or dense (C%32==0)      -> v8h vector loads, no LDS
//   vec4  : conv with C%4==0                 -> LDS offset table + v4h loads
//   scalar: conv with odd C (C=3,10)         -> LDS offset table + d16 loads
// Padded-K A values are garbage * 0-weight (weights zero-padded), so no
// predication is needed in the hot loop.  nt==2 -> 16x32 tile per wave.
// ---------------------------------------------------------------------------
struct ConvP {
  const h16* x; const h16* wp; const float* bias; const float* alpha;
  h16* y16; float* y32;
  int N, C, H, W, O, KH, KW, OH, OW, Kpad, Opad, prelu, nt, contig, vec4;
};

__global__ __launch_bounds__(128) void wmma_conv_kernel(ConvP p) {
  extern __shared__ int xoff[];                 // [Kpad] gather offsets
  const int C = p.C;
  const int K = C * p.KH * p.KW;

  if (!p.contig) {                              // build offset table once/block
    for (int k = threadIdx.x; k < p.Kpad; k += 128) {
      int off = 0;
      if (k < K) {
        const int c  = k % C;                   // K order: (ky,kx,c)
        const int s  = k / C;
        const int ky = s / p.KW;
        const int kx = s - ky * p.KW;
        off = (ky * p.W + kx) * C + c;          // NHWC relative offset
      }
      xoff[k] = off;
    }
    __syncthreads();
  }

  const int wave = threadIdx.x >> 5;
  const int lane = threadIdx.x & 31;
  const long M  = (long)p.N * p.OH * p.OW;
  const long m0 = ((long)blockIdx.x * 4 + wave) * 16;
  if (m0 >= M) return;                          // wave-uniform: EXEC all-1s
  const int hi  = (lane >> 4) & 1;              // lower / upper half-wave
  const int HWo = p.OH * p.OW;

  // A-fragment row geometry (lane = row M=lane%16 per ISA layout)
  long pix = m0 + (lane & 15);
  if (pix >= M) pix = M - 1;                    // clamp: loads stay in-bounds
  const h16* xin;
  if (p.contig) {
    xin = p.x + pix * C;                        // 1x1 / dense: row contiguous
  } else {
    const int nb  = (int)(pix / HWo);
    const int rem = (int)(pix - (long)nb * HWo);
    const int oy  = rem / p.OW;
    const int ox  = rem - oy * p.OW;
    xin = p.x + ((long)(nb * p.H + oy) * p.W + ox) * C;   // NHWC pixel base
  }

  // B-fragment column geometry (lane = column N=lane%16, halves = ISA 8-K groups)
  const int ocBase = blockIdx.y * (16 * p.nt) + (lane & 15);
  const h16* w0 = p.wp + (long)ocBase * p.Kpad + hi * 8;
  const h16* w1 = w0 + (long)16 * p.Kpad;       // only deref'd when nt==2

  v8f acc0 = {}, acc1 = {};
  for (int kb = 0; kb < p.Kpad; kb += 32) {
    if (kb + 32 < p.Kpad)
      __builtin_prefetch((const void*)(w0 + kb + 32), 0, 1);  // global_prefetch_b8
    v16h a;
    if (p.contig) {                             // fully contiguous A row
#pragma unroll
      for (int h2 = 0; h2 < 2; ++h2) {
        const v8h ch = *(const v8h*)(xin + kb + h2 * 16 + hi * 8);
#pragma unroll
        for (int j = 0; j < 8; ++j) a[h2 * 8 + j] = ch[j];
      }
    } else if (p.vec4) {                        // c-contiguous 4-vectors
#pragma unroll
      for (int h2 = 0; h2 < 2; ++h2) {
        const int kc = kb + h2 * 16 + hi * 8;   // ISA 16-bit A layout K mapping
        const v4h c0 = *(const v4h*)(xin + xoff[kc]);
        const v4h c1 = *(const v4h*)(xin + xoff[kc + 4]);
#pragma unroll
        for (int j = 0; j < 4; ++j) { a[h2 * 8 + j] = c0[j]; a[h2 * 8 + 4 + j] = c1[j]; }
      }
    } else {                                    // scalar gather (C=3, C=10)
#pragma unroll
      for (int h2 = 0; h2 < 2; ++h2) {
        const int kc = kb + h2 * 16 + hi * 8;
        const int4 oA = *(const int4*)&xoff[kc];
        const int4 oB = *(const int4*)&xoff[kc + 4];
        a[h2 * 8 + 0] = xin[oA.x]; a[h2 * 8 + 1] = xin[oA.y];
        a[h2 * 8 + 2] = xin[oA.z]; a[h2 * 8 + 3] = xin[oA.w];
        a[h2 * 8 + 4] = xin[oB.x]; a[h2 * 8 + 5] = xin[oB.y];
        a[h2 * 8 + 6] = xin[oB.z]; a[h2 * 8 + 7] = xin[oB.w];
      }
    }
    const v16h b0 = __builtin_shufflevector(*(const v8h*)(w0 + kb), *(const v8h*)(w0 + kb + 16), 0, 1, 2, 3, 4, 5, 6, 7, 8, 9, 10, 11, 12, 13, 14, 15);    // two 16B chunks per lane
    acc0 = __builtin_amdgcn_wmma_f32_16x16x32_f16(
        false, a, false, b0, (short)0, acc0, false, false);
    if (p.nt == 2) {
      const v16h b1 = __builtin_shufflevector(*(const v8h*)(w1 + kb), *(const v8h*)(w1 + kb + 16), 0, 1, 2, 3, 4, 5, 6, 7, 8, 9, 10, 11, 12, 13, 14, 15);
      acc1 = __builtin_amdgcn_wmma_f32_16x16x32_f16(
          false, a, false, b1, (short)0, acc1, false, false);
    }
  }

#pragma unroll
  for (int t = 0; t < 2; ++t) {
    if (t >= p.nt) break;
    const int oc = ocBase + t * 16;
    if (oc >= p.O) continue;
    const v8f acc = t ? acc1 : acc0;
    const float bi = p.bias[oc];
    const float al = p.prelu ? p.alpha[oc] : 0.f;
#pragma unroll
    for (int v = 0; v < 8; ++v) {               // C/D layout: row = v + 8*hi
      const long pv = m0 + v + hi * 8;
      if (pv >= M) break;
      float val = acc[v] + bi;
      if (p.prelu && val < 0.f) val *= al;
      if (p.y32) {                              // heads: NCHW f32 (ref layout)
        const int nb2 = (int)(pv / HWo);
        const int r2  = (int)(pv - (long)nb2 * HWo);
        p.y32[((long)nb2 * p.O + oc) * HWo + r2] = val;
      } else {                                  // activations: NHWC f16
        p.y16[pv * p.O + oc] = (h16)val;        // coalesced across lanes
      }
    }
  }
}

// ---------------------------------------------------------------------------
// Support kernels (all activations NHWC f16)
// ---------------------------------------------------------------------------
__global__ void nchw_to_nhwc_f16_kernel(const float* __restrict__ x, h16* __restrict__ y,
                                        int N, int C, long HW) {
  long i = (long)blockIdx.x * blockDim.x + threadIdx.x;   // over N*HW*C
  if (i >= (long)N * HW * C) return;
  int  c  = (int)(i % C);
  long t  = i / C;
  long hw = t % HW;
  int  n  = (int)(t / HW);
  y[i] = (h16)x[((long)n * C + c) * HW + hw];
}

// OIHW f32 -> f16 [Opad][Kpad], K ordered (ky,kx,c)
__global__ void pack_w_kernel(const float* __restrict__ w, h16* __restrict__ wp,
                              int O, int C, int KH, int KW, int Kpad, long tot) {
  long i = (long)blockIdx.x * blockDim.x + threadIdx.x;
  if (i >= tot) return;
  const int K = C * KH * KW;
  int o = (int)(i / Kpad);
  int k = (int)(i - (long)o * Kpad);
  h16 v = (h16)0.f;
  if (o < O && k < K) {
    const int c  = k % C;
    const int s  = k / C;
    const int ky = s / KW;
    const int kx = s - ky * KW;
    v = (h16)w[(((long)o * C + c) * KH + ky) * KW + kx];
  }
  wp[i] = v;
}

__global__ void maxpool_kernel(const h16* __restrict__ x, h16* __restrict__ y,
                               int N, int C, int H, int W, int OH, int OW, int k, int s) {
  long i = (long)blockIdx.x * blockDim.x + threadIdx.x;   // NHWC, c innermost
  long tot = (long)N * OH * OW * C;
  if (i >= tot) return;
  int c  = (int)(i % C); long t = i / C;
  int ow = (int)(t % OW); t /= OW;
  int oh = (int)(t % OH);
  int n  = (int)(t / OH);
  int h0 = oh * s, w0 = ow * s;
  int h1 = min(h0 + k, H), w1 = min(w0 + k, W);   // ceil-mode: -inf pad == clamp
  const h16* xp = x + (long)n * H * W * C + c;
  float m = -3.0e38f;
  for (int hh = h0; hh < h1; ++hh)
    for (int ww = w0; ww < w1; ++ww)
      m = fmaxf(m, (float)xp[((long)hh * W + ww) * C]);
  y[i] = (h16)m;
}

// NHWC -> [N, W*H*C]  (matches torch permute(0,3,2,1).reshape of NCHW)
__global__ void flatten_kernel(const h16* __restrict__ x, h16* __restrict__ y,
                               int N, int C, int Hc, int Wc) {
  const int K = C * Hc * Wc;
  long i = (long)blockIdx.x * blockDim.x + threadIdx.x;
  if (i >= (long)N * K) return;
  int n = (int)(i / K);
  int k = (int)(i - (long)n * K);
  int w = k / (Hc * C);
  int r = k - w * (Hc * C);
  int h = r / C;
  int c = r - h * C;
  y[i] = x[(((long)n * Hc + h) * Wc + w) * C + c];
}

// softmax over 2 channels; layout [B, 2, HW]. HW=1 covers [N,2] row softmax.
__global__ void softmax2_kernel(const float* __restrict__ z, float* __restrict__ out,
                                int B, long HW) {
  long i = (long)blockIdx.x * blockDim.x + threadIdx.x;
  if (i >= (long)B * HW) return;
  long b = i / HW, p = i - b * HW;
  float z0 = z[(b * 2 + 0) * HW + p];
  float z1 = z[(b * 2 + 1) * HW + p];
  float m = fmaxf(z0, z1);
  float e0 = __expf(z0 - m), e1 = __expf(z1 - m);
  float s = e0 + e1;
  out[(b * 2 + 0) * HW + p] = e0 / s;
  out[(b * 2 + 1) * HW + p] = e1 / s;
}

// ---------------------------------------------------------------------------
// Host helpers
// ---------------------------------------------------------------------------
static inline int rup16(int x) { return (x + 15) & ~15; }
static inline int rup32(int x) { return (x + 31) & ~31; }
static inline unsigned cdivu(long a, long b) { return (unsigned)((a + b - 1) / b); }

static void launch_conv(hipStream_t s, const h16* x, const h16* wp,
                        const float* bias, const float* alpha,
                        h16* y16, float* y32,
                        int N, int C, int H, int W, int O, int KH, int KW) {
  ConvP p;
  p.x = x; p.wp = wp; p.bias = bias; p.alpha = alpha; p.y16 = y16; p.y32 = y32;
  p.N = N; p.C = C; p.H = H; p.W = W; p.O = O; p.KH = KH; p.KW = KW;
  p.OH = H - KH + 1; p.OW = W - KW + 1;
  p.Kpad = rup32(C * KH * KW); p.Opad = rup16(O);
  p.prelu  = (alpha != nullptr) ? 1 : 0;
  p.nt     = (p.Opad % 32 == 0) ? 2 : 1;
  p.contig = (KH == 1 && KW == 1 && (C % 32 == 0)) ? 1 : 0;   // K == Kpad too
  p.vec4   = (!p.contig && (C % 4 == 0)) ? 1 : 0;
  long M = (long)N * p.OH * p.OW;
  dim3 g(cdivu(M, 64), (unsigned)(p.Opad / (16 * p.nt)));
  size_t shmem = p.contig ? 0 : (size_t)p.Kpad * sizeof(int);
  wmma_conv_kernel<<<g, dim3(128), shmem, s>>>(p);
}

static void launch_pool(hipStream_t s, const h16* x, h16* y,
                        int N, int C, int H, int W, int k, int st,
                        int* OHo, int* OWo) {
  int OH = (H - k + st - 1) / st + 1;   // ceil((H-k)/s)+1
  int OW = (W - k + st - 1) / st + 1;
  long tot = (long)N * OH * OW * C;
  maxpool_kernel<<<cdivu(tot, 256), 256, 0, s>>>(x, y, N, C, H, W, OH, OW, k, st);
  *OHo = OH; *OWo = OW;
}

static h16* pack_w(hipStream_t s, const float* w, h16** cursor,
                   int O, int C, int KH, int KW) {
  int Opad = rup16(O), Kpad = rup32(C * KH * KW);
  h16* dst = *cursor;
  *cursor += (size_t)Opad * Kpad;
  long tot = (long)Opad * Kpad;
  pack_w_kernel<<<cdivu(tot, 256), 256, 0, s>>>(w, dst, O, C, KH, KW, Kpad, tot);
  return dst;
}

// ---------------------------------------------------------------------------
extern "C" void kernel_launch(void* const* d_in, const int* in_sizes, int n_in,
                              void* d_out, int out_size, void* d_ws, size_t ws_size,
                              hipStream_t stream) {
  (void)in_sizes; (void)n_in; (void)out_size; (void)ws_size;
  const float* imgs    = (const float*)d_in[0];
  const float* crops24 = (const float*)d_in[1];
  const float* crops48 = (const float*)d_in[2];
  const float *pw1=(const float*)d_in[3],  *pb1=(const float*)d_in[4],  *pa1=(const float*)d_in[5];
  const float *pw2=(const float*)d_in[6],  *pb2=(const float*)d_in[7],  *pa2=(const float*)d_in[8];
  const float *pw3=(const float*)d_in[9],  *pb3=(const float*)d_in[10], *pa3=(const float*)d_in[11];
  const float *pw41=(const float*)d_in[12],*pb41=(const float*)d_in[13];
  const float *pw42=(const float*)d_in[14],*pb42=(const float*)d_in[15];
  const float *rw1=(const float*)d_in[16], *rb1=(const float*)d_in[17], *ra1=(const float*)d_in[18];
  const float *rw2=(const float*)d_in[19], *rb2=(const float*)d_in[20], *ra2=(const float*)d_in[21];
  const float *rw3=(const float*)d_in[22], *rb3=(const float*)d_in[23], *ra3=(const float*)d_in[24];
  const float *rd4w=(const float*)d_in[25],*rd4b=(const float*)d_in[26],*ra4=(const float*)d_in[27];
  const float *rd51w=(const float*)d_in[28],*rd51b=(const float*)d_in[29];
  const float *rd52w=(const float*)d_in[30],*rd52b=(const float*)d_in[31];
  const float *ow1=(const float*)d_in[32], *ob1=(const float*)d_in[33], *oa1=(const float*)d_in[34];
  const float *ow2=(const float*)d_in[35], *ob2=(const float*)d_in[36], *oa2=(const float*)d_in[37];
  const float *ow3=(const float*)d_in[38], *ob3=(const float*)d_in[39], *oa3=(const float*)d_in[40];
  const float *ow4=(const float*)d_in[41], *ob4=(const float*)d_in[42], *oa4=(const float*)d_in[43];
  const float *od5w=(const float*)d_in[44],*od5b=(const float*)d_in[45],*oa5=(const float*)d_in[46];
  const float *od61w=(const float*)d_in[47],*od61b=(const float*)d_in[48];
  const float *od62w=(const float*)d_in[49],*od62b=(const float*)d_in[50];
  const float *od63w=(const float*)d_in[51],*od63b=(const float*)d_in[52];

  char* ws = (char*)d_ws;
  const size_t BUF = (size_t)84 << 20;        // 84 MiB per ping-pong f16 buffer
  h16* b0 = (h16*)ws;
  h16* b1 = (h16*)(ws + BUF);
  h16* wcur = (h16*)(ws + 2 * BUF);

  // pack all weight matrices to f16 [Opad][Kpad], K ordered (ky,kx,c)
  h16* Wp1  = pack_w(stream, pw1,  &wcur, 10, 3, 3, 3);
  h16* Wp2  = pack_w(stream, pw2,  &wcur, 16, 10, 3, 3);
  h16* Wp3  = pack_w(stream, pw3,  &wcur, 32, 16, 3, 3);
  h16* Wp41 = pack_w(stream, pw41, &wcur, 2, 32, 1, 1);
  h16* Wp42 = pack_w(stream, pw42, &wcur, 4, 32, 1, 1);
  h16* Wr1  = pack_w(stream, rw1,  &wcur, 28, 3, 3, 3);
  h16* Wr2  = pack_w(stream, rw2,  &wcur, 48, 28, 3, 3);
  h16* Wr3  = pack_w(stream, rw3,  &wcur, 64, 48, 2, 2);
  h16* Wrd4 = pack_w(stream, rd4w, &wcur, 128, 576, 1, 1);
  h16* Wr51 = pack_w(stream, rd51w,&wcur, 2, 128, 1, 1);
  h16* Wr52 = pack_w(stream, rd52w,&wcur, 4, 128, 1, 1);
  h16* Wo1  = pack_w(stream, ow1,  &wcur, 32, 3, 3, 3);
  h16* Wo2  = pack_w(stream, ow2,  &wcur, 64, 32, 3, 3);
  h16* Wo3  = pack_w(stream, ow3,  &wcur, 64, 64, 3, 3);
  h16* Wo4  = pack_w(stream, ow4,  &wcur, 128, 64, 2, 2);
  h16* Wod5 = pack_w(stream, od5w, &wcur, 256, 1152, 1, 1);
  h16* Wo61 = pack_w(stream, od61w,&wcur, 2, 256, 1, 1);
  h16* Wo62 = pack_w(stream, od62w,&wcur, 4, 256, 1, 1);
  h16* Wo63 = pack_w(stream, od63w,&wcur, 10, 256, 1, 1);

  float* out    = (float*)d_out;
  float* p_reg  = out;                   // [8,4,355,355]  = 4032800
  float* p_prob = out + 4032800;         // [8,2,355,355]  = 2016400
  float* r_b    = out + 6049200;         // [2048,4]
  float* r_a    = r_b + 8192;            // [2048,2]
  float* o_b    = r_a + 4096;            // [512,4]
  float* o_c    = o_b + 2048;            // [512,10]
  float* o_a    = o_c + 5120;            // [512,2]

  int ph, pw;

  // ---------------- PNet ----------------
  {
    long n = 8L * 3 * 720 * 720;
    nchw_to_nhwc_f16_kernel<<<cdivu(n, 256), 256, 0, stream>>>(imgs, b0, 8, 3, 720L * 720);
    launch_conv(stream, b0, Wp1, pb1, pa1, b1, nullptr, 8, 3, 720, 720, 10, 3, 3);   // 718
    launch_pool(stream, b1, b0, 8, 10, 718, 718, 2, 2, &ph, &pw);                    // 359
    launch_conv(stream, b0, Wp2, pb2, pa2, b1, nullptr, 8, 10, ph, pw, 16, 3, 3);    // 357
    launch_conv(stream, b1, Wp3, pb3, pa3, b0, nullptr, 8, 16, 357, 357, 32, 3, 3);  // 355
    launch_conv(stream, b0, Wp42, pb42, nullptr, nullptr, p_reg, 8, 32, 355, 355, 4, 1, 1);
    float* plog = (float*)b1;
    launch_conv(stream, b0, Wp41, pb41, nullptr, nullptr, plog, 8, 32, 355, 355, 2, 1, 1);
    long hw = 355L * 355;
    softmax2_kernel<<<cdivu(8 * hw, 256), 256, 0, stream>>>(plog, p_prob, 8, hw);
  }

  // ---------------- RNet ----------------
  {
    long n = 2048L * 3 * 24 * 24;
    nchw_to_nhwc_f16_kernel<<<cdivu(n, 256), 256, 0, stream>>>(crops24, b0, 2048, 3, 24L * 24);
    launch_conv(stream, b0, Wr1, rb1, ra1, b1, nullptr, 2048, 3, 24, 24, 28, 3, 3);  // 22
    launch_pool(stream, b1, b0, 2048, 28, 22, 22, 3, 2, &ph, &pw);                   // 11
    launch_conv(stream, b0, Wr2, rb2, ra2, b1, nullptr, 2048, 28, ph, pw, 48, 3, 3); // 9
    launch_pool(stream, b1, b0, 2048, 48, 9, 9, 3, 2, &ph, &pw);                     // 4
    launch_conv(stream, b0, Wr3, rb3, ra3, b1, nullptr, 2048, 48, ph, pw, 64, 2, 2); // 3
    long nf = 2048L * 576;
    flatten_kernel<<<cdivu(nf, 256), 256, 0, stream>>>(b1, b0, 2048, 64, 3, 3);
    launch_conv(stream, b0, Wrd4, rd4b, ra4, b1, nullptr, 2048, 576, 1, 1, 128, 1, 1);
    launch_conv(stream, b1, Wr52, rd52b, nullptr, nullptr, r_b, 2048, 128, 1, 1, 4, 1, 1);
    float* rlog = (float*)b0;
    launch_conv(stream, b1, Wr51, rd51b, nullptr, nullptr, rlog, 2048, 128, 1, 1, 2, 1, 1);
    softmax2_kernel<<<cdivu(2048, 256), 256, 0, stream>>>(rlog, r_a, 2048, 1);
  }

  // ---------------- ONet ----------------
  {
    long n = 512L * 3 * 48 * 48;
    nchw_to_nhwc_f16_kernel<<<cdivu(n, 256), 256, 0, stream>>>(crops48, b0, 512, 3, 48L * 48);
    launch_conv(stream, b0, Wo1, ob1, oa1, b1, nullptr, 512, 3, 48, 48, 32, 3, 3);   // 46
    launch_pool(stream, b1, b0, 512, 32, 46, 46, 3, 2, &ph, &pw);                    // 23
    launch_conv(stream, b0, Wo2, ob2, oa2, b1, nullptr, 512, 32, ph, pw, 64, 3, 3);  // 21
    launch_pool(stream, b1, b0, 512, 64, 21, 21, 3, 2, &ph, &pw);                    // 10
    launch_conv(stream, b0, Wo3, ob3, oa3, b1, nullptr, 512, 64, ph, pw, 64, 3, 3);  // 8
    launch_pool(stream, b1, b0, 512, 64, 8, 8, 2, 2, &ph, &pw);                      // 4
    launch_conv(stream, b0, Wo4, ob4, oa4, b1, nullptr, 512, 64, ph, pw, 128, 2, 2); // 3
    long nf = 512L * 1152;
    flatten_kernel<<<cdivu(nf, 256), 256, 0, stream>>>(b1, b0, 512, 128, 3, 3);
    launch_conv(stream, b0, Wod5, od5b, oa5, b1, nullptr, 512, 1152, 1, 1, 256, 1, 1);
    launch_conv(stream, b1, Wo62, od62b, nullptr, nullptr, o_b, 512, 256, 1, 1, 4, 1, 1);
    launch_conv(stream, b1, Wo63, od63b, nullptr, nullptr, o_c, 512, 256, 1, 1, 10, 1, 1);
    float* olog = (float*)b0;
    launch_conv(stream, b1, Wo61, od61b, nullptr, nullptr, olog, 512, 256, 1, 1, 2, 1, 1);
    softmax2_kernel<<<cdivu(512, 256), 256, 0, stream>>>(olog, o_a, 512, 1);
  }
}
